// Update_Failed_78726750535838
// MI455X (gfx1250) — matrix operand order repaired and checked by reading, not compiled, not run on hardware
//
#include <hip/hip_runtime.h>
#include <cstdint>
#include <cstddef>

// ---------------------------------------------------------------------------
// Problem constants (from the reference)
// ---------------------------------------------------------------------------
#define NTOK   4096
#define DIMF   384
#define CORRIN 882
#define CORRKP 896     // CORRIN padded to a multiple of BK (zero-filled)
#define GKK    768
#define GIJ    96

typedef __attribute__((ext_vector_type(16))) _Float16 v16h;
typedef __attribute__((ext_vector_type(8)))  _Float16 v8h;
typedef __attribute__((ext_vector_type(8)))  float    v8f;

// ---------------------------------------------------------------------------
// WMMA GEMM with CDNA5 async global->LDS staging (double buffered).
//   Y[N,M] (f32, + optional f16 shadow) =
//       act( X16[N,K] (optionally row-gathered) @ W16[M,K]^T + bias ) (+res)
// Block: 256 threads (8 wave32), tile 128x128, K-step 32, f16 WMMA, f32 acc.
// K must be a multiple of 32 (inputs are pre-padded), N,M multiples of 128.
// ---------------------------------------------------------------------------
#define BM 128
#define BN 128
#define BK 32
#define APAD 8   // pad rows to 40 halves = 80B (16B aligned)

__device__ __forceinline__ void async_cp_b128(unsigned lds_off, const _Float16* src) {
    // GLOBAL_LOAD_ASYNC_TO_LDS_B128: per-lane 16B global -> LDS, ASYNCcnt.
    asm volatile("global_load_async_to_lds_b128 %0, %1, off"
                 :: "v"(lds_off), "v"(src) : "memory");
}

__global__ __launch_bounds__(256)
void gemm_wmma_k(const _Float16* __restrict__ X, const int* __restrict__ gather,
                 const _Float16* __restrict__ W, const float* __restrict__ bias,
                 const float* __restrict__ resAdd,
                 float* __restrict__ Y, _Float16* __restrict__ Y16,
                 int N, int K, int M, int act)
{
    __shared__ alignas(16) _Float16 As[2][BM][BK + APAD];
    __shared__ alignas(16) _Float16 Bs[2][BN][BK + APAD];

    const int tid     = threadIdx.x;
    const int rowBase = blockIdx.y * BM;
    const int colBase = blockIdx.x * BN;
    const int wid  = tid >> 5;
    const int lane = tid & 31;
    const int wm = wid & 3;        // 4 waves along M (32 rows each)
    const int wn = wid >> 2;       // 2 waves along N (64 cols each)
    const int grp = lane >> 4;     // half-wave group
    const int l16 = lane & 15;

    // Per-lane async staging assignments: 512 granules of 16B per tile
    // (128 rows x 4), each lane owns 2 for A and 2 for B.
    const _Float16* aSrc[2]; unsigned aDst[2];
    const _Float16* bSrc[2]; unsigned bDst[2];
#pragma unroll
    for (int j = 0; j < 2; ++j) {
        int g = tid + j * 256;
        int r = g >> 2;            // row in tile
        int c = (g & 3) * 8;       // halve offset within row (16B granule)
        int xr = gather ? gather[rowBase + r] : (rowBase + r);
        aSrc[j] = X + (size_t)xr * K + c;
        aDst[j] = (unsigned)(uintptr_t)&As[0][r][c];   // LDS aperture: low 32b = LDS addr
        bSrc[j] = W + (size_t)(colBase + r) * K + c;
        bDst[j] = (unsigned)(uintptr_t)&Bs[0][r][c];
    }
    const unsigned bufStride = (unsigned)sizeof(As[0]);  // == sizeof(Bs[0])

    auto issue = [&](int buf, int k0) {
        unsigned boff = (unsigned)buf * bufStride;
#pragma unroll
        for (int j = 0; j < 2; ++j) async_cp_b128(aDst[j] + boff, aSrc[j] + k0);
#pragma unroll
        for (int j = 0; j < 2; ++j) async_cp_b128(bDst[j] + boff, bSrc[j] + k0);
    };

    v8f acc[2][4];
#pragma unroll
    for (int tm = 0; tm < 2; ++tm)
#pragma unroll
        for (int tn = 0; tn < 4; ++tn)
#pragma unroll
            for (int r = 0; r < 8; ++r) acc[tm][tn][r] = 0.0f;

    const int ksteps = K / BK;
    int cur = 0;
    issue(0, 0);
    for (int ks = 0; ks < ksteps; ++ks) {
        if (ks + 1 < ksteps) {
            issue(cur ^ 1, (ks + 1) * BK);                       // prefetch next tile
            asm volatile("s_wait_asynccnt 0x4" ::: "memory");    // stage ks done (in-order)
        } else {
            asm volatile("s_wait_asynccnt 0x0" ::: "memory");
        }
        __syncthreads();

        union V16 { v16h v; v8h h[2]; };
        v16h afrag[2], bfrag[4];
#pragma unroll
        for (int tm = 0; tm < 2; ++tm) {
            const _Float16* ap = &As[cur][wm * 32 + tm * 16 + l16][0];
            V16 t;
            t.h[0] = *(const v8h*)(ap + grp * 8);        // K 0-7  / 8-15
            t.h[1] = *(const v8h*)(ap + 16 + grp * 8);   // K16-23 / 24-31
            afrag[tm] = t.v;
        }
#pragma unroll
        for (int tn = 0; tn < 4; ++tn) {
            const _Float16* bp = &Bs[cur][wn * 64 + tn * 16 + l16][0];
            V16 t;
            t.h[0] = *(const v8h*)(bp + grp * 8);
            t.h[1] = *(const v8h*)(bp + 16 + grp * 8);
            bfrag[tn] = t.v;
        }
#pragma unroll
        for (int tm = 0; tm < 2; ++tm)
#pragma unroll
            for (int tn = 0; tn < 4; ++tn)
                acc[tm][tn] = __builtin_amdgcn_wmma_f32_16x16x32_f16(
                    false, afrag[tm], false, bfrag[tn],
                    (short)0, acc[tm][tn], false, false);

        __syncthreads();    // all waves done reading `cur` before it is refilled
        cur ^= 1;
    }

    // ---- epilogue: bias + act + residual, dual f32/f16 store ----
#pragma unroll
    for (int tm = 0; tm < 2; ++tm)
#pragma unroll
        for (int tn = 0; tn < 4; ++tn) {
            int col = colBase + wn * 64 + tn * 16 + l16;
#pragma unroll
            for (int r = 0; r < 8; ++r) {
                int row = rowBase + wm * 32 + tm * 16 + grp * 8 + r;
                if (row < N && col < M) {
                    float v = acc[tm][tn][r] + (bias ? bias[col] : 0.0f);
                    if (act == 1)      v = fmaxf(v, 0.0f);
                    else if (act == 2) v = 1.0f / (1.0f + expf(-v));
                    if (resAdd) v += resAdd[(size_t)row * M + col];
                    Y[(size_t)row * M + col] = v;
                    if (Y16) Y16[(size_t)row * M + col] = (_Float16)v;
                }
            }
        }
}

// ---------------------------------------------------------------------------
// Conversion kernels (f32 -> f16 shadows for the async WMMA path)
// ---------------------------------------------------------------------------
__global__ void cvt_k(const float* __restrict__ s, _Float16* __restrict__ d, int n) {
    int i = blockIdx.x * 256 + threadIdx.x;
    if (i < n) d[i] = (_Float16)s[i];
}

// dst[m, kp] = (k < K) ? src[m, k] : 0   (row-major, K padded to KP)
__global__ void cvtpad_k(const float* __restrict__ s, _Float16* __restrict__ d,
                         int M, int K, int KP) {
    int i = blockIdx.x * 256 + threadIdx.x;
    if (i < M * KP) {
        int m = i / KP, k = i % KP;
        d[i] = (_Float16)((k < K) ? s[(size_t)m * K + k] : 0.0f);
    }
}

// dst[n, kp] = (k < K) ? src[k, n] : 0   (feature-major -> row-major, padded)
__global__ void tcvt_k(const float* __restrict__ s, _Float16* __restrict__ d,
                       int N, int K, int KP) {
    int i = blockIdx.x * 256 + threadIdx.x;
    if (i < N * KP) {
        int n = i / KP, k = i % KP;
        d[i] = (_Float16)((k < K) ? s[(size_t)k * N + n] : 0.0f);
    }
}

// ---------------------------------------------------------------------------
// Elementwise / reduction kernels
// ---------------------------------------------------------------------------
__global__ void fill_k(float* p, float v, int n) {
    int i = blockIdx.x * 256 + threadIdx.x;
    if (i < n) p[i] = v;
}

__global__ void iisum_k(const int* ii, float* out, int n) {
    __shared__ float sh[256];
    float s = 0.0f;
    for (int i = threadIdx.x; i < n; i += 256) s += (float)ii[i];
    sh[threadIdx.x] = s;
    __syncthreads();
    for (int off = 128; off; off >>= 1) {
        if (threadIdx.x < off) sh[threadIdx.x] += sh[threadIdx.x + off];
        __syncthreads();
    }
    if (threadIdx.x == 0) out[0] = sh[0] * 1e-10f;
}

// out[n,d] = netF[d,n] + inpF[d,n] + c[n,d] + iibias
__global__ void fuse_k(const float* __restrict__ netF, const float* __restrict__ inpF,
                       const float* __restrict__ c, const float* __restrict__ iibias,
                       float* __restrict__ out, int N, int D)
{
    int i = blockIdx.x * 256 + threadIdx.x;
    if (i < N * D) {
        int n = i / D, d = i % D;
        out[i] = netF[(size_t)d * N + n] + inpF[(size_t)d * N + n] + c[i] + iibias[0];
    }
}

// LayerNorm over last dim (wave per row), optional ReLU + optional f16 shadow.
__global__ __launch_bounds__(256)
void ln_k(float* __restrict__ x, const float* __restrict__ g,
          const float* __restrict__ b, int N, int D, int relu,
          _Float16* __restrict__ out16)
{
    int row  = blockIdx.x * (blockDim.x >> 5) + (threadIdx.x >> 5);
    int lane = threadIdx.x & 31;
    if (row >= N) return;
    float* r = x + (size_t)row * D;
    float s = 0.0f, ss = 0.0f;
    for (int e = lane; e < D; e += 32) { float v = r[e]; s += v; ss += v * v; }
    for (int off = 16; off; off >>= 1) {
        s  += __shfl_xor(s,  off, 32);
        ss += __shfl_xor(ss, off, 32);
    }
    float m   = s / (float)D;
    float var = ss / (float)D - m * m;
    float inv = rsqrtf(var + 1e-3f);
    for (int e = lane; e < D; e += 32) {
        float v = (r[e] - m) * inv * g[e] + b[e];
        if (relu) v = fmaxf(v, 0.0f);
        r[e] = v;
        if (out16) out16[(size_t)row * D + e] = (_Float16)v;
    }
}

// Per-token prev/next neighbor in same kk group ordered by jj.
__global__ __launch_bounds__(256)
void neighbors_k(const int* __restrict__ jj, const int* __restrict__ kk,
                 int* __restrict__ ix, int* __restrict__ jx, int N)
{
    __shared__ int pv[256], pi[256], nv[256], ni[256];
    int n = blockIdx.x;
    int myk = kk[n], myj = jj[n];
    int bpv = -1, bpi = N;
    int bnv = 0x7fffffff, bni = N;
    for (int j = threadIdx.x; j < N; j += 256) {
        int same = (kk[j] == myk);
        int vj = jj[j];
        int pvj = (same && vj < myj) ? vj : 0;
        int nvj = (same && vj > myj) ? vj : N;
        if (pvj > bpv || (pvj == bpv && j < bpi)) { bpv = pvj; bpi = j; }
        if (nvj < bnv || (nvj == bnv && j < bni)) { bnv = nvj; bni = j; }
    }
    pv[threadIdx.x] = bpv; pi[threadIdx.x] = bpi;
    nv[threadIdx.x] = bnv; ni[threadIdx.x] = bni;
    __syncthreads();
    for (int off = 128; off; off >>= 1) {
        if (threadIdx.x < off) {
            int t = threadIdx.x, o = t + off;
            if (pv[o] > pv[t] || (pv[o] == pv[t] && pi[o] < pi[t])) { pv[t] = pv[o]; pi[t] = pi[o]; }
            if (nv[o] < nv[t] || (nv[o] == nv[t] && ni[o] < ni[t])) { nv[t] = nv[o]; ni[t] = ni[o]; }
        }
        __syncthreads();
    }
    if (threadIdx.x == 0) { ix[n] = pi[0]; jx[n] = ni[0]; }
}

__device__ inline void atomicMaxF(float* a, float v) {
    int* ai = (int*)a;
    int old = *ai;
    while (__int_as_float(old) < v) {
        int assumed = old;
        old = atomicCAS(ai, assumed, __float_as_int(v));
        if (old == assumed) break;
    }
}

__global__ void segmax_k(const float* __restrict__ gx, const int* __restrict__ map,
                         float* __restrict__ gmax, int N, int D)
{
    int i = blockIdx.x * 256 + threadIdx.x;
    if (i < N * D) {
        int n = i / D, d = i % D;
        atomicMaxF(&gmax[(size_t)map[n] * D + d], gx[i]);
    }
}

__global__ void expsum_k(float* __restrict__ gx, const int* __restrict__ map,
                         const float* __restrict__ gmax, float* __restrict__ den,
                         int N, int D)
{
    int i = blockIdx.x * 256 + threadIdx.x;
    if (i < N * D) {
        int n = i / D, d = i % D;
        int g = map[n];
        float e = expf(gx[i] - gmax[(size_t)g * D + d]);
        gx[i] = e;
        atomicAdd(&den[(size_t)g * D + d], e);
    }
}

__global__ void wsum_k(const float* __restrict__ fx, const float* __restrict__ e,
                       const int* __restrict__ map, const float* __restrict__ den,
                       float* __restrict__ ysum, int N, int D)
{
    int i = blockIdx.x * 256 + threadIdx.x;
    if (i < N * D) {
        int n = i / D, d = i % D;
        int g = map[n];
        float w = e[i] / den[(size_t)g * D + d];
        atomicAdd(&ysum[(size_t)g * D + d], fx[i] * w);
    }
}

__global__ void gated_add_k(const float* __restrict__ x, const float* __restrict__ gate,
                            const float* __restrict__ res, float* __restrict__ out, int n)
{
    int i = blockIdx.x * 256 + threadIdx.x;
    if (i < n) out[i] = x[i] + gate[i] * res[i];
}

__global__ __launch_bounds__(256)
void heads_k(const float* __restrict__ net, const int* __restrict__ ii,
             const float* __restrict__ dW, const float* __restrict__ db,
             const float* __restrict__ wW, const float* __restrict__ wb,
             float* __restrict__ dout, float* __restrict__ wout, int N, int D)
{
    int t    = blockIdx.x * (blockDim.x >> 5) + (threadIdx.x >> 5);
    int lane = threadIdx.x & 31;
    if (t >= N) return;
    const float* x = net + (size_t)t * D;
    float s0 = 0, s1 = 0, s2 = 0, s3 = 0;
    for (int e = lane; e < D; e += 32) {
        float h = fmaxf(x[e], 0.0f);
        s0 += h * dW[e];     s1 += h * dW[D + e];
        s2 += h * wW[e];     s3 += h * wW[D + e];
    }
    for (int off = 16; off; off >>= 1) {
        s0 += __shfl_down(s0, off, 32);
        s1 += __shfl_down(s1, off, 32);
        s2 += __shfl_down(s2, off, 32);
        s3 += __shfl_down(s3, off, 32);
    }
    if (lane == 0) {
        float ib = (float)ii[t] * 1e-10f;
        dout[(size_t)t * 2 + 0] = s0 + db[0] + ib;
        dout[(size_t)t * 2 + 1] = s1 + db[1] + ib;
        wout[(size_t)t * 2 + 0] = 1.0f / (1.0f + expf(-(s2 + wb[0]))) + ib;
        wout[(size_t)t * 2 + 1] = 1.0f / (1.0f + expf(-(s3 + wb[1]))) + ib;
    }
}

// ---------------------------------------------------------------------------
// Host orchestration
// ---------------------------------------------------------------------------
static inline void launch_gemm(hipStream_t s, const _Float16* X, const int* gather,
                               const _Float16* Wt, const float* bias,
                               const float* res, float* Y, _Float16* Y16,
                               int N, int K, int M, int act)
{
    dim3 grid(M / BN, N / BM);
    gemm_wmma_k<<<grid, dim3(256), 0, s>>>(X, gather, Wt, bias, res, Y, Y16, N, K, M, act);
}

extern "C" void kernel_launch(void* const* d_in, const int* in_sizes, int n_in,
                              void* d_out, int out_size, void* d_ws, size_t ws_size,
                              hipStream_t stream)
{
    (void)in_sizes; (void)n_in; (void)out_size; (void)ws_size;

    const float* netF  = (const float*)d_in[0];   // [384,4096] feature-major
    const float* inpF  = (const float*)d_in[1];
    const float* corrF = (const float*)d_in[2];   // [882,4096]
    const int* ii     = (const int*)d_in[4];
    const int* jj     = (const int*)d_in[5];
    const int* kk     = (const int*)d_in[6];
    const int* kk_map = (const int*)d_in[7];
    const int* ij_map = (const int*)d_in[8];

    const int p = 11;
    auto P = [&](int i) { return (const float*)d_in[p + i]; };
    const float *corr0W = P(0),  *corr0b = P(1);
    const float *corr1W = P(2),  *corr1b = P(3);
    const float *clnG   = P(4),  *clnB   = P(5);
    const float *corr2W = P(6),  *corr2b = P(7);
    const float *normG  = P(8),  *normB  = P(9);
    const float *c10W = P(10), *c10b = P(11);
    const float *c11W = P(12), *c11b = P(13);
    const float *c20W = P(14), *c20b = P(15);
    const float *c21W = P(16), *c21b = P(17);
    const float *akfW = P(18), *akfb = P(19);
    const float *akgW = P(20), *akgb = P(21);
    const float *akhW = P(22), *akhb = P(23);
    const float *aifW = P(24), *aifb = P(25);
    const float *aigW = P(26), *aigb = P(27);
    const float *aihW = P(28), *aihb = P(29);
    const float *ln1G = P(30), *ln1B = P(31);
    const float *g1W  = P(32), *g1b  = P(33);
    const float *r10W = P(34), *r10b = P(35);
    const float *r11W = P(36), *r11b = P(37);
    const float *ln2G = P(38), *ln2B = P(39);
    const float *g2W  = P(40), *g2b  = P(41);
    const float *r20W = P(42), *r20b = P(43);
    const float *r21W = P(44), *r21b = P(45);
    const float *dW = P(46), *db = P(47);
    const float *wW = P(48), *wb = P(49);

    // ---------------- workspace carve ----------------
    const size_t ND = (size_t)NTOK * DIMF;
    float* ws = (float*)d_ws;
    size_t off = 0;
    float* t1   = ws + off; off += ND;
    float* t2   = ws + off; off += ND;
    float* nx   = ws + off; off += ND;
    float* ny   = ws + off; off += ND;
    float* gmax = ws + off; off += (size_t)GKK * DIMF;
    float* den  = ws + off; off += (size_t)GKK * DIMF;
    float* ysum = ws + off; off += (size_t)GKK * DIMF;
    float* iib  = ws + off; off += 4;                 // keep 16B alignment
    int*   ix   = (int*)(ws + off); off += NTOK;
    int*   jx   = (int*)(ws + off); off += NTOK;

    _Float16* h16 = (_Float16*)(ws + off);
    size_t ho = 0;
    _Float16* hT  = h16 + ho; ho += ND;                          // shadow of t1/t2
    _Float16* hN  = h16 + ho; ho += ND;                          // shadow of net
    _Float16* hY  = h16 + ho; ho += (size_t)GKK * DIMF;          // shadow of ysum
    _Float16* cT  = h16 + ho; ho += (size_t)NTOK * CORRKP;       // corr, transposed+padded
    _Float16* wc0 = h16 + ho; ho += (size_t)DIMF * CORRKP;
    const int NW = 18;
    _Float16* w16[NW];
    for (int i = 0; i < NW; ++i) { w16[i] = h16 + ho; ho += (size_t)DIMF * DIMF; }
    const float* wsrc[NW] = { corr1W, corr2W, c10W, c11W, c20W, c21W,
                              akfW, akgW, akhW, aifW, aigW, aihW,
                              g1W, r10W, r11W, g2W, r20W, r21W };
    _Float16 *wcorr1 = w16[0], *wcorr2 = w16[1], *wc10 = w16[2], *wc11 = w16[3],
             *wc20 = w16[4], *wc21 = w16[5], *wakf = w16[6], *wakg = w16[7],
             *wakh = w16[8], *waif = w16[9], *waig = w16[10], *waih = w16[11],
             *wg1 = w16[12], *wr10 = w16[13], *wr11 = w16[14], *wg2 = w16[15],
             *wr20 = w16[16], *wr21 = w16[17];

    float* onet = (float*)d_out;
    float* od   = onet + ND;
    float* ow   = od + (size_t)NTOK * 2;

    const int NDi = NTOK * DIMF;
    const int EB  = (NDi + 255) / 256;
    const int LNB = NTOK / 8;
    const int WB  = (DIMF * DIMF + 255) / 256;

    // 0) f16 shadows of the weights (per call; deterministic)
    cvtpad_k<<<(DIMF * CORRKP + 255) / 256, 256, 0, stream>>>(corr0W, wc0, DIMF, CORRIN, CORRKP);
    for (int i = 0; i < NW; ++i)
        cvt_k<<<WB, 256, 0, stream>>>(wsrc[i], w16[i], DIMF * DIMF);
    tcvt_k<<<(NTOK * CORRKP + 255) / 256, 256, 0, stream>>>(corrF, cT, NTOK, CORRIN, CORRKP);

    // 1) dummy ii bias
    iisum_k<<<1, 256, 0, stream>>>(ii, iib, NTOK);

    // 2) corr MLP: Lin-ReLU-Lin-LN-ReLU-Lin
    launch_gemm(stream, cT, nullptr, wc0, corr0b, nullptr, t1, hT, NTOK, CORRKP, DIMF, 1);
    launch_gemm(stream, hT, nullptr, wcorr1, corr1b, nullptr, t2, nullptr, NTOK, DIMF, DIMF, 0);
    ln_k<<<LNB, 256, 0, stream>>>(t2, clnG, clnB, NTOK, DIMF, 1, hT);
    launch_gemm(stream, hT, nullptr, wcorr2, corr2b, nullptr, t1, nullptr, NTOK, DIMF, DIMF, 0);

    // 3) net = LN(net + inp + c + ii_bias)
    fuse_k<<<EB, 256, 0, stream>>>(netF, inpF, t1, iib, nx, NTOK, DIMF);
    ln_k<<<LNB, 256, 0, stream>>>(nx, normG, normB, NTOK, DIMF, 0, hN);

    // 4) neighbor indices
    neighbors_k<<<NTOK, 256, 0, stream>>>(jj, kk, ix, jx, NTOK);

    // 5) net += mlp2(net[ix]); net += mlp2(net[jx])
    launch_gemm(stream, hN, ix, wc10, c10b, nullptr, t1, hT, NTOK, DIMF, DIMF, 1);
    launch_gemm(stream, hT, nullptr, wc11, c11b, nx, ny, hN, NTOK, DIMF, DIMF, 0);
    { float* t = nx; nx = ny; ny = t; }
    launch_gemm(stream, hN, jx, wc20, c20b, nullptr, t1, hT, NTOK, DIMF, DIMF, 1);
    launch_gemm(stream, hT, nullptr, wc21, c21b, nx, ny, hN, NTOK, DIMF, DIMF, 0);
    { float* t = nx; nx = ny; ny = t; }

    // 6) two scatter-softmax aggregations
    struct AggP { const int* map; int G;
                  const _Float16 *fW, *gW, *hW;
                  const float *fb, *gb, *hb; };
    AggP aggs[2] = {
        { kk_map, GKK, wakf, wakg, wakh, akfb, akgb, akhb },
        { ij_map, GIJ, waif, waig, waih, aifb, aigb, aihb },
    };
    for (int a = 0; a < 2; ++a) {
        const AggP& A = aggs[a];
        const int GD = A.G * DIMF;
        const int GB = (GD + 255) / 256;
        launch_gemm(stream, hN, nullptr, A.fW, A.fb, nullptr, t1, nullptr, NTOK, DIMF, DIMF, 0); // fx
        launch_gemm(stream, hN, nullptr, A.gW, A.gb, nullptr, t2, nullptr, NTOK, DIMF, DIMF, 0); // gx
        fill_k<<<GB, 256, 0, stream>>>(gmax, -3.0e38f, GD);
        segmax_k<<<EB, 256, 0, stream>>>(t2, A.map, gmax, NTOK, DIMF);
        fill_k<<<GB, 256, 0, stream>>>(den, 0.0f, GD);
        expsum_k<<<EB, 256, 0, stream>>>(t2, A.map, gmax, den, NTOK, DIMF);
        fill_k<<<GB, 256, 0, stream>>>(ysum, 0.0f, GD);
        wsum_k<<<EB, 256, 0, stream>>>(t1, t2, A.map, den, ysum, NTOK, DIMF);
        cvt_k<<<GB, 256, 0, stream>>>(ysum, hY, GD);
        launch_gemm(stream, hY, A.map, A.hW, A.hb, nx, ny, hN, NTOK, DIMF, DIMF, 0);
        { float* t = nx; nx = ny; ny = t; }
    }

    // 7) LN -> gated residual -> LN -> gated residual (final into d_out)
    ln_k<<<LNB, 256, 0, stream>>>(nx, ln1G, ln1B, NTOK, DIMF, 0, hN);
    launch_gemm(stream, hN, nullptr, wg1,  g1b,  nullptr, t1, nullptr, NTOK, DIMF, DIMF, 2);
    launch_gemm(stream, hN, nullptr, wr10, r10b, nullptr, t2, hT, NTOK, DIMF, DIMF, 1);
    launch_gemm(stream, hT, nullptr, wr11, r11b, nullptr, ny, nullptr, NTOK, DIMF, DIMF, 0);
    gated_add_k<<<EB, 256, 0, stream>>>(nx, t1, ny, nx, NDi);

    ln_k<<<LNB, 256, 0, stream>>>(nx, ln2G, ln2B, NTOK, DIMF, 0, hN);
    launch_gemm(stream, hN, nullptr, wg2,  g2b,  nullptr, t1, nullptr, NTOK, DIMF, DIMF, 2);
    launch_gemm(stream, hN, nullptr, wr20, r20b, nullptr, t2, hT, NTOK, DIMF, DIMF, 1);
    launch_gemm(stream, hT, nullptr, wr21, r21b, nullptr, ny, nullptr, NTOK, DIMF, DIMF, 0);
    gated_add_k<<<EB, 256, 0, stream>>>(nx, t1, ny, onet, NDi);

    // 8) d / w heads
    heads_k<<<NTOK / 8, 256, 0, stream>>>(onet, ii, dW, db, wW, wb, od, ow, NTOK, DIMF);
}
